// GAT_40295383171393
// MI455X (gfx1250) — matrix operand order repaired and checked by reading, not compiled, not run on hardware
//
#include <hip/hip_runtime.h>
#include <hip/hip_bf16.h>
#include <math.h>

typedef long long i64;
typedef __attribute__((ext_vector_type(16))) __bf16 v16bf;
typedef __attribute__((ext_vector_type(8)))  __bf16 v8bf;
typedef __attribute__((ext_vector_type(8)))  float  v8f;
typedef __attribute__((ext_vector_type(4)))  unsigned int u32x4;
typedef __attribute__((ext_vector_type(8)))  int i32x8;
typedef __attribute__((ext_vector_type(4)))  int i32x4;

#define N_NODES   20000
#define N_EDGES   320000
#define IN_CH     256
#define HID       128
#define HEADS     4
#define OUT_CH    200
#define NUM_GRAPHS 64
#define NEG_SLOPE 0.2f

// ---------------------------------------------------------------------------
// utility kernels
// ---------------------------------------------------------------------------
__global__ void fill_f32(float* p, float v, int n) {
    int i = blockIdx.x * blockDim.x + threadIdx.x;
    if (i < n) p[i] = v;
}

__global__ void f32_to_bf16(const float* __restrict__ in, __bf16* __restrict__ out, int n) {
    int i = blockIdx.x * blockDim.x + threadIdx.x;
    if (i < n) out[i] = (__bf16)in[i];
}

// W [K,M] f32 row-major  ->  Wt [M,K] bf16 (each WMMA B-fragment becomes a
// contiguous 16-element run per lane)
__global__ void transpose_f32_to_bf16(const float* __restrict__ W, __bf16* __restrict__ Wt,
                                      int K, int M) {
    __shared__ float tile[16][17];
    int kb = blockIdx.x * 16, mb = blockIdx.y * 16;
    int tx = threadIdx.x & 15, ty = threadIdx.x >> 4;   // 256 threads = 16x16
    int k = kb + ty, m = mb + tx;
    if (k < K && m < M) tile[ty][tx] = W[(size_t)k * M + m];
    __syncthreads();
    int k2 = kb + tx, m2 = mb + ty;
    if (k2 < K && m2 < M) Wt[(size_t)m2 * K + k2] = (__bf16)tile[tx][ty];
}

// ---------------------------------------------------------------------------
// bf16 WMMA GEMM:  C[N,M] = A[N,K] @ B[K,M]   (B given as Bt[M,K] bf16)
//
// Block = 256 threads = 8 waves; block computes a 128x64 output macro-tile.
// The 64-column bf16 weight strip (64 x K, <=64KB) is DMA'd into LDS once per
// block by the Tensor Data Mover (wave 0), then each wave register-blocks a
// 16x64 strip: 4 f32 accumulators. Per K-step: 1 A fragment (global), then
// ALL 4 B fragments (LDS) are fetched before the 4 independent WMMAs issue,
// so ds loads clause together and WMMAs run back-to-back.
// ---------------------------------------------------------------------------
__global__ __launch_bounds__(256)
void gemm_bf16_wmma(const __bf16* __restrict__ A, const __bf16* __restrict__ Bt,
                    float* __restrict__ C, int N, int K, int M) {
    extern __shared__ __bf16 Bs[];                 // [64][K]
    const int lane  = threadIdx.x & 31;
    const int wave  = threadIdx.x >> 5;            // 0..7 -> row tile within block
    const int tilesN = N >> 4;                     // 16-row tiles
    const int colBlocks = M >> 6;                  // 64-col macro tiles
    const int rb = blockIdx.x / colBlocks;
    const int cb = blockIdx.x % colBlocks;
    const int col0 = cb << 6;

    // ---- stage B strip (Bt rows col0..col0+63, full K) into LDS via TDM ----
    if (wave == 0) {
        unsigned long long gaddr =
            (unsigned long long)(uintptr_t)(Bt + (size_t)col0 * K);
        unsigned int lds_off = (unsigned int)(uintptr_t)&Bs[0]; // LDS aperture low bits
        u32x4 g0 = {};
        g0[0] = 1u;                                      // count=1 (valid), user mode
        g0[1] = lds_off;                                 // lds_addr (bytes)
        g0[2] = (unsigned int)(gaddr & 0xFFFFFFFFu);     // global_addr[31:0]
        g0[3] = (unsigned int)((gaddr >> 32) & 0x1FFFFFFu) | (2u << 30); // addr[56:32], type=2
        i32x8 g1 = {};
        g1[0] = (1 << 16);                 // data_size = 1 -> 2 bytes/elem
        g1[1] = (K & 0xFFFF) << 16;        // tensor_dim0[15:0]  (bits 63:48)
        g1[2] = ((K >> 16) & 0xFFFF) | (64 << 16); // tensor_dim0 hi | tensor_dim1 lo
        g1[3] = (K << 16);                 // tensor_dim1 hi (0) | tile_dim0 = K
        g1[4] = 64;                        // tile_dim1 = 64 rows, tile_dim2 = 0
        g1[5] = K;                         // tensor_dim0_stride[31:0]
        g1[6] = 0;                         // stride hi | tensor_dim1_stride lo
        g1[7] = 0;
        i32x4 z4 = {};
        i32x8 z8 = {};
        __builtin_amdgcn_tensor_load_to_lds(g0, g1, z4, z4, z8, 0);
        __builtin_amdgcn_s_wait_tensorcnt(0);
    }
    __syncthreads();                                  // publish LDS tile

    const int tm = rb * 8 + wave;                     // this wave's 16-row tile
    if (tm >= tilesN) return;                         // wave-uniform, after barrier

    const bool hi = lane >= 16;
    const int  l  = lane & 15;
    const __bf16* Arow = A + (size_t)(tm * 16 + l) * K;
    // per-column-tile LDS base for this lane (column t*16 + l)
    const __bf16* Bl0 = Bs + (size_t)(0 * 16 + l) * K;
    const __bf16* Bl1 = Bs + (size_t)(1 * 16 + l) * K;
    const __bf16* Bl2 = Bs + (size_t)(2 * 16 + l) * K;
    const __bf16* Bl3 = Bs + (size_t)(3 * 16 + l) * K;

    v8f acc0 = {}, acc1 = {}, acc2 = {}, acc3 = {};
    for (int kk = 0; kk < K; kk += 32) {
        // A fragment: lane<16 -> K = kk..kk+15 ; lane>=16 -> kk+16..kk+31
        v16bf a;
        const int aoff = kk + (hi ? 16 : 0);
        *((v8bf*)&a)     = *(const v8bf*)(Arow + aoff);
        *((v8bf*)&a + 1) = *(const v8bf*)(Arow + aoff + 8);
        // all 4 B fragments first (8x ds_load_b128 clause, single wait), then
        // 4 independent WMMAs back-to-back
        const int boff = kk + (hi ? 16 : 0);
        v16bf b0, b1, b2, b3;
        *((v8bf*)&b0)     = *(const v8bf*)(Bl0 + boff);
        *((v8bf*)&b0 + 1) = *(const v8bf*)(Bl0 + boff + 8);
        *((v8bf*)&b1)     = *(const v8bf*)(Bl1 + boff);
        *((v8bf*)&b1 + 1) = *(const v8bf*)(Bl1 + boff + 8);
        *((v8bf*)&b2)     = *(const v8bf*)(Bl2 + boff);
        *((v8bf*)&b2 + 1) = *(const v8bf*)(Bl2 + boff + 8);
        *((v8bf*)&b3)     = *(const v8bf*)(Bl3 + boff);
        *((v8bf*)&b3 + 1) = *(const v8bf*)(Bl3 + boff + 8);
        acc0 = __builtin_amdgcn_wmma_f32_16x16x32_bf16(false, a, false, b0, (short)0, acc0, false, false);
        acc1 = __builtin_amdgcn_wmma_f32_16x16x32_bf16(false, a, false, b1, (short)0, acc1, false, false);
        acc2 = __builtin_amdgcn_wmma_f32_16x16x32_bf16(false, a, false, b2, (short)0, acc2, false, false);
        acc3 = __builtin_amdgcn_wmma_f32_16x16x32_bf16(false, a, false, b3, (short)0, acc3, false, false);
    }
    // C/D layout: VGPR r -> row tm*16 + r (+8 for high lanes), col tile + l
    const int rbase = tm * 16 + (hi ? 8 : 0);
    v8f accs[4] = {acc0, acc1, acc2, acc3};
    #pragma unroll
    for (int t = 0; t < 4; ++t) {
        const int col = col0 + t * 16 + l;
        #pragma unroll
        for (int r = 0; r < 8; ++r)
            C[(size_t)(rbase + r) * M + col] = accs[t][r];
    }
}

// ---------------------------------------------------------------------------
// attention coefficients: alpha_s[n,h] = <h[n,h,:], a_src[h,:]>  (wave per n,h)
// ---------------------------------------------------------------------------
__global__ void attn_alpha(const float* __restrict__ h,
                           const float* __restrict__ a_src, const float* __restrict__ a_dst,
                           float* __restrict__ alpha_s, float* __restrict__ alpha_d,
                           int N, int H, int C) {
    int wid  = (blockIdx.x * blockDim.x + threadIdx.x) >> 5;
    int lane = threadIdx.x & 31;
    if (wid >= N * H) return;
    int n = wid / H, hh = wid % H;
    const float* hp = h + (size_t)n * H * C + (size_t)hh * C;
    const float* as = a_src + hh * C;
    const float* ad = a_dst + hh * C;
    float ss = 0.f, sd = 0.f;
    for (int c = lane; c < C; c += 32) { float v = hp[c]; ss += v * as[c]; sd += v * ad[c]; }
    for (int off = 16; off > 0; off >>= 1) {
        ss += __shfl_down(ss, off, 32);
        sd += __shfl_down(sd, off, 32);
    }
    if (lane == 0) { alpha_s[wid] = ss; alpha_d[wid] = sd; }
}

// float atomic max via monotonic int/uint trick (native integer atomics)
__device__ __forceinline__ void atomicMaxF32(float* addr, float val) {
    if (val >= 0.0f) atomicMax((int*)addr, __float_as_int(val));
    else             atomicMin((unsigned int*)addr, (unsigned int)__float_as_int(val));
}

// leaky-relu edge logits + segment max over dst
__global__ void edge_logits_max(const i64* __restrict__ ei, int E, int N,
                                const float* __restrict__ alpha_s,
                                const float* __restrict__ alpha_d,
                                float* __restrict__ eval, float* __restrict__ mx, int H) {
    int e = blockIdx.x * blockDim.x + threadIdx.x;
    int Etot = E + N;
    if (e >= Etot) return;
    int s = (e < E) ? (int)ei[e]     : (e - E);
    int d = (e < E) ? (int)ei[E + e] : (e - E);
    for (int hh = 0; hh < H; ++hh) {
        float v = alpha_s[s * H + hh] + alpha_d[d * H + hh];
        v = (v > 0.f) ? v : NEG_SLOPE * v;
        eval[(size_t)e * H + hh] = v;
        atomicMaxF32(&mx[d * H + hh], v);
    }
}

// exp(e - max) + segment sum of denominators
__global__ void edge_exp(const i64* __restrict__ ei, int E, int N,
                         const float* __restrict__ mx,
                         float* __restrict__ eval, float* __restrict__ denom, int H) {
    int e = blockIdx.x * blockDim.x + threadIdx.x;
    int Etot = E + N;
    if (e >= Etot) return;
    int d = (e < E) ? (int)ei[E + e] : (e - E);
    for (int hh = 0; hh < H; ++hh) {
        float v = expf(eval[(size_t)e * H + hh] - mx[d * H + hh]);
        eval[(size_t)e * H + hh] = v;
        atomicAdd(&denom[d * H + hh], v);
    }
}

// message scatter: out[dst] += alpha * h[src]   (one wave per edge, float4 lanes)
__global__ void edge_scatter(const i64* __restrict__ ei, int E, int N,
                             const float* __restrict__ h,
                             const float* __restrict__ eval,
                             const float* __restrict__ denom,
                             float* __restrict__ out, int H, int C) {
    int wid  = (blockIdx.x * blockDim.x + threadIdx.x) >> 5;
    int lane = threadIdx.x & 31;
    int Etot = E + N;
    if (wid >= Etot) return;
    int e = wid;
    int s = (e < E) ? (int)ei[e]     : (e - E);
    int d = (e < E) ? (int)ei[E + e] : (e - E);
    int HC = H * C;
    const float* hs = h   + (size_t)s * HC;
    float*       od = out + (size_t)d * HC;
    for (int base = lane * 4; base < HC; base += 128) {
        int hh = base / C;
        float al = eval[(size_t)e * H + hh] / (denom[d * H + hh] + 1e-16f);
        float4 v = *(const float4*)(hs + base);
        atomicAdd(&od[base + 0], v.x * al);
        atomicAdd(&od[base + 1], v.y * al);
        atomicAdd(&od[base + 2], v.z * al);
        atomicAdd(&od[base + 3], v.w * al);
    }
}

// out = elu(out + bias[col])
__global__ void bias_elu(float* __restrict__ x, const float* __restrict__ bias, int N, int M) {
    int i = blockIdx.x * blockDim.x + threadIdx.x;
    if (i >= N * M) return;
    float v = x[i] + bias[i % M];
    x[i] = (v > 0.f) ? v : (expf(v) - 1.f);
}

// global mean pool: accumulate sums/counts per graph (wave per node)
__global__ void pool_sum(const float* __restrict__ h, const i64* __restrict__ batch,
                         float* __restrict__ sums, float* __restrict__ cnts, int N, int C) {
    int wid  = (blockIdx.x * blockDim.x + threadIdx.x) >> 5;
    int lane = threadIdx.x & 31;
    if (wid >= N) return;
    int g = (int)batch[wid];
    const float* hp = h + (size_t)wid * C;
    for (int c = lane; c < C; c += 32) atomicAdd(&sums[g * C + c], hp[c]);
    if (lane == 0) atomicAdd(&cnts[g], 1.0f);
}

// logits[g,o] = (sums[g,:]/cnt) @ fc_w + fc_b
__global__ void fc_out(const float* __restrict__ sums, const float* __restrict__ cnts,
                       const float* __restrict__ W, const float* __restrict__ b,
                       float* __restrict__ out, int G, int C, int O) {
    int g = blockIdx.x, o = threadIdx.x;
    if (g >= G || o >= O) return;
    float inv = 1.0f / fmaxf(cnts[g], 1.0f);
    float acc = 0.f;
    for (int c = 0; c < C; ++c) acc += sums[g * C + c] * inv * W[c * O + o];
    out[g * O + o] = acc + b[o];
}

// ---------------------------------------------------------------------------
// host side
// ---------------------------------------------------------------------------
static inline int ceil_div(int a, int b) { return (a + b - 1) / b; }

extern "C" void kernel_launch(void* const* d_in, const int* in_sizes, int n_in,
                              void* d_out, int out_size, void* d_ws, size_t ws_size,
                              hipStream_t stream) {
    const float* x      = (const float*)d_in[0];
    const i64*   ei     = (const i64*)d_in[1];
    const i64*   batch  = (const i64*)d_in[2];
    const float* W1     = (const float*)d_in[3];
    const float* as1    = (const float*)d_in[4];
    const float* ad1    = (const float*)d_in[5];
    const float* b1     = (const float*)d_in[6];
    const float* W2     = (const float*)d_in[7];
    const float* as2    = (const float*)d_in[8];
    const float* ad2    = (const float*)d_in[9];
    const float* b2     = (const float*)d_in[10];
    const float* W3     = (const float*)d_in[11];
    const float* as3    = (const float*)d_in[12];
    const float* ad3    = (const float*)d_in[13];
    const float* b3     = (const float*)d_in[14];
    const float* fcw    = (const float*)d_in[15];
    const float* fcb    = (const float*)d_in[16];
    float* out = (float*)d_out;

    // bump allocator over workspace (256B aligned)
    char* wsp = (char*)d_ws;
    auto alloc = [&](size_t bytes) -> void* {
        void* p = (void*)wsp;
        wsp += (bytes + 255) & ~(size_t)255;
        return p;
    };
    const int N = N_NODES, E = N_EDGES, Etot = N_EDGES + N_NODES;
    float*  hbuf0  = (float*) alloc((size_t)N * 512 * 4);   // projection h
    float*  hbuf1  = (float*) alloc((size_t)N * 512 * 4);   // aggregated out / next input
    __bf16* xbf    = (__bf16*)alloc((size_t)N * 512 * 2);
    __bf16* wtbf   = (__bf16*)alloc((size_t)512 * 512 * 2);
    float*  al_s   = (float*) alloc((size_t)N * HEADS * 4);
    float*  al_d   = (float*) alloc((size_t)N * HEADS * 4);
    float*  mx     = (float*) alloc((size_t)N * HEADS * 4);
    float*  denom  = (float*) alloc((size_t)N * HEADS * 4);
    float*  evals  = (float*) alloc((size_t)Etot * HEADS * 4);
    float*  sums   = (float*) alloc((size_t)NUM_GRAPHS * HID * 4);
    float*  cnts   = (float*) alloc((size_t)NUM_GRAPHS * 4);

    auto run_layer = [&](const float* cur, int K, int H, int C,
                         const float* W, const float* a_s, const float* a_d,
                         const float* bias, float* hproj, float* oagg) {
        const int M = H * C;
        // 1) input -> bf16
        {
            int n = N * K;
            f32_to_bf16<<<ceil_div(n, 256), 256, 0, stream>>>(cur, xbf, n);
        }
        // 2) W -> bf16 transposed
        {
            dim3 g(ceil_div(K, 16), ceil_div(M, 16));
            transpose_f32_to_bf16<<<g, 256, 0, stream>>>(W, wtbf, K, M);
        }
        // 3) h = x @ W  (TDM-staged LDS weights + WMMA)
        {
            int rowBlocks = ceil_div(N / 16, 8);   // 8 row-tiles per block
            int colBlocks = M / 64;                // 4 col-tiles per block
            size_t lds = (size_t)64 * K * sizeof(__bf16);
            gemm_bf16_wmma<<<rowBlocks * colBlocks, 256, lds, stream>>>(
                xbf, wtbf, hproj, N, K, M);
        }
        // 4) attention dot products
        {
            int waves = N * H;
            attn_alpha<<<ceil_div(waves, 8), 256, 0, stream>>>(hproj, a_s, a_d, al_s, al_d, N, H, C);
        }
        // 5) init segment buffers
        fill_f32<<<ceil_div(N * H, 256), 256, 0, stream>>>(mx,   -INFINITY, N * H);
        fill_f32<<<ceil_div(N * H, 256), 256, 0, stream>>>(denom, 0.0f,     N * H);
        fill_f32<<<ceil_div(N * M, 256), 256, 0, stream>>>(oagg,  0.0f,     N * M);
        // 6) edge logits + segment max
        edge_logits_max<<<ceil_div(Etot, 256), 256, 0, stream>>>(ei, E, N, al_s, al_d, evals, mx, H);
        // 7) exp + segment sum
        edge_exp<<<ceil_div(Etot, 256), 256, 0, stream>>>(ei, E, N, mx, evals, denom, H);
        // 8) weighted scatter
        edge_scatter<<<ceil_div(Etot, 8), 256, 0, stream>>>(ei, E, N, hproj, evals, denom, oagg, H, C);
        // 9) bias + ELU
        bias_elu<<<ceil_div(N * M, 256), 256, 0, stream>>>(oagg, bias, N, M);
    };

    // layer 1: [N,256] -> [N,512]
    run_layer(x,     IN_CH,       HEADS, HID, W1, as1, ad1, b1, hbuf0, hbuf1);
    // layer 2: [N,512] -> [N,512]
    run_layer(hbuf1, HEADS * HID, HEADS, HID, W2, as2, ad2, b2, hbuf0, hbuf1);
    // layer 3: [N,512] -> [N,128]
    run_layer(hbuf1, HEADS * HID, 1,     HID, W3, as3, ad3, b3, hbuf0, hbuf1);

    // mean pool + FC
    fill_f32<<<ceil_div(NUM_GRAPHS * HID, 256), 256, 0, stream>>>(sums, 0.0f, NUM_GRAPHS * HID);
    fill_f32<<<1, 256, 0, stream>>>(cnts, 0.0f, NUM_GRAPHS);
    pool_sum<<<ceil_div(N, 8), 256, 0, stream>>>(hbuf1, batch, sums, cnts, N, HID);
    fc_out<<<NUM_GRAPHS, 256, 0, stream>>>(sums, cnts, fcw, fcb, out, NUM_GRAPHS, HID, OUT_CH);
}
